// Network_10797547782358
// MI455X (gfx1250) — matrix operand order repaired and checked by reading, not compiled, not run on hardware
//
#include <hip/hip_runtime.h>
#include <hip/hip_bf16.h>
#include <math.h>

typedef __attribute__((ext_vector_type(16))) __bf16 v16bf;
typedef __attribute__((ext_vector_type(8)))  __bf16 v8bf;
typedef __attribute__((ext_vector_type(8)))  float  v8f;

#define HH    76
#define WW    120
#define HWSZ  9120          // 76*120
#define AA    9
#define CIN   1024
#define CMID  512
#define NANCH 82080         // 9*76*120
#define TOPN  300
#define NCLS  21
#define FCD   2048
#define KFC   50176         // 1024*7*7
#define NEGINF (-3.4e38f)

static __device__ __forceinline__ __bf16 f2bf(float f) {
  unsigned u = __builtin_bit_cast(unsigned, f);
  unsigned r = (u + 0x7FFFu + ((u >> 16) & 1u)) >> 16;   // RNE
  unsigned short s = (unsigned short)r;
  return __builtin_bit_cast(__bf16, s);
}

static __device__ __forceinline__ v16bf pack16(v8bf lo, v8bf hi) {
  return __builtin_shufflevector(lo, hi, 0, 1, 2, 3, 4, 5, 6, 7,
                                 8, 9, 10, 11, 12, 13, 14, 15);
}

// ---------------------------------------------------------------------------
// 0a. repack rpn_w (512,1024,3,3) fp32 -> bf16 wArs[9][512][1024] (tap-major)
// ---------------------------------------------------------------------------
__global__ void prep_conv_weights(const float* __restrict__ w, __bf16* __restrict__ wArs) {
  int idx = blockIdx.x * 256 + threadIdx.x;
  if (idx >= 9 * CMID * CIN) return;
  int rs = idx / (CMID * CIN), rem = idx % (CMID * CIN);
  int m = rem / CIN, c = rem % CIN;
  int r = rs / 3, s = rs % 3;
  wArs[idx] = f2bf(w[((m * CIN + c) * 3 + r) * 3 + s]);
}

// ---------------------------------------------------------------------------
// 0b. transpose feature map fp32 [c][pos] -> bf16 featT[pos][c]
//     gives contiguous-K vector loads for the conv B fragments
// ---------------------------------------------------------------------------
__global__ void prep_featT(const float* __restrict__ feat, __bf16* __restrict__ featT) {
  int idx = blockIdx.x * 256 + threadIdx.x;
  if (idx >= CIN * HWSZ) return;
  int c = idx / HWSZ, pos = idx % HWSZ;              // coalesced reads
  featT[(size_t)pos * CIN + c] = f2bf(feat[idx]);
}

// ---------------------------------------------------------------------------
// 1. RPN 3x3 conv = 9 shifted GEMMs (tap loop outer, channel-K inner).
//    Border predicate + position are loop-invariant per tap; B fragment is one
//    32B v16bf load, A fragment two 16B v8bf loads. 288 WMMA per wave-tile.
// ---------------------------------------------------------------------------
__global__ void __launch_bounds__(32) conv3x3_wmma(
    const __bf16* __restrict__ featT,  // [9120][1024]
    const __bf16* __restrict__ wArs,   // [9][512][1024]
    const float* __restrict__ bias,    // [512]
    __bf16* __restrict__ outbf)        // [512][9120]
{
  const int nt = blockIdx.x, mt = blockIdx.y;
  const int lane = threadIdx.x & 31;
  const int half = lane >> 4, l15 = lane & 15;
  const int n = nt * 16 + l15;
  const int y = n / WW, x = n % WW;
  const int m = mt * 16 + l15;
  const v16bf zv = {};
  v8f acc = {};
#pragma unroll
  for (int rs = 0; rs < 9; ++rs) {
    const int r = rs / 3, s = rs % 3;
    const int yy = y + r - 1, xx = x + s - 1;
    const bool valid = (yy >= 0) & (yy < HH) & (xx >= 0) & (xx < WW);
    const int posn = valid ? (yy * WW + xx) : 0;
    const __bf16* __restrict__ bbase = featT + (size_t)posn * CIN + 16 * half;
    const __bf16* __restrict__ abase = wArs + ((size_t)rs * CMID + m) * CIN + 16 * half;
    for (int c0 = 0; c0 < CIN; c0 += 32) {
      v8bf a0 = *(const v8bf*)(abase + c0);
      v8bf a1 = *(const v8bf*)(abase + c0 + 8);
      v16bf a = pack16(a0, a1);
      v16bf b = *(const v16bf*)(bbase + c0);
      b = valid ? b : zv;
      acc = __builtin_amdgcn_wmma_f32_16x16x32_bf16(false, a, false, b, (short)0, acc, false, false);
    }
  }
#pragma unroll
  for (int r = 0; r < 8; ++r) {
    int Mg = mt * 16 + r + 8 * half;
    int Ng = nt * 16 + l15;
    float v = acc[r] + bias[Mg];
    outbf[Mg * HWSZ + Ng] = f2bf(v > 0.f ? v : 0.f);
  }
}

// ---------------------------------------------------------------------------
// 2. generic 16x16-tile WMMA GEMM (used for the small GEMMs)
// ---------------------------------------------------------------------------
template <int AF32, int BF32, int BIASROW>
__global__ void __launch_bounds__(32) gemm16_wmma(
    const void* __restrict__ Ap, int lda,
    const void* __restrict__ Bp, int ldb,
    const float* __restrict__ bias,
    float* __restrict__ out, int ldo,
    int M, int N, int K)
{
  const int nt = blockIdx.x, mt = blockIdx.y;
  const int lane = threadIdx.x & 31;
  const int half = lane >> 4, l15 = lane & 15;
  const int n = nt * 16 + l15;
  const int m = mt * 16 + l15;
  const float*  Af = (const float*)Ap;
  const __bf16* Ab = (const __bf16*)Ap;
  const float*  Bf = (const float*)Bp;
  const __bf16* Bb = (const __bf16*)Bp;
  const __bf16 bz = f2bf(0.f);
  v8f acc = {};
  for (int k0 = 0; k0 < K; k0 += 32) {
    v16bf a, b;
#pragma unroll
    for (int e = 0; e < 16; ++e) {
      int ka = k0 + e + 16 * half;
      if (m < M && ka < K) a[e] = AF32 ? f2bf(Af[m * lda + ka]) : Ab[m * lda + ka];
      else                 a[e] = bz;
      int kb = k0 + e + 16 * half;
      if (n < N && kb < K) b[e] = BF32 ? f2bf(Bf[kb * ldb + n]) : Bb[kb * ldb + n];
      else                 b[e] = bz;
    }
    acc = __builtin_amdgcn_wmma_f32_16x16x32_bf16(false, a, false, b, (short)0, acc, false, false);
  }
#pragma unroll
  for (int r = 0; r < 8; ++r) {
    int Mg = mt * 16 + r + 8 * half, Ng = nt * 16 + l15;
    if (Mg < M && Ng < N)
      out[Mg * ldo + Ng] = acc[r] + (BIASROW ? bias[Mg] : bias[Ng]);
  }
}

// ---------------------------------------------------------------------------
// 3. proposal decode
// ---------------------------------------------------------------------------
__global__ void proposal_kernel(const float* __restrict__ cls_s,   // [18][9120]
                                const float* __restrict__ bbox_p,  // [36][9120]
                                const float* __restrict__ anchors, // [82080][4]
                                float* __restrict__ scores,
                                float* __restrict__ boxes)
{
  int j = blockIdx.x * 256 + threadIdx.x;
  if (j >= NANCH) return;
  int pos = j / AA, a = j % AA;
  scores[j] = cls_s[(AA + a) * HWSZ + pos] - cls_s[a * HWSZ + pos];
  float d0 = bbox_p[(4 * a + 0) * HWSZ + pos];
  float d1 = bbox_p[(4 * a + 1) * HWSZ + pos];
  float d2 = bbox_p[(4 * a + 2) * HWSZ + pos];
  float d3 = bbox_p[(4 * a + 3) * HWSZ + pos];
  float a0 = anchors[j * 4 + 0], a1 = anchors[j * 4 + 1];
  float a2 = anchors[j * 4 + 2], a3 = anchors[j * 4 + 3];
  float aw = a2 - a0 + 1.f, ah = a3 - a1 + 1.f;
  float acx = a0 + 0.5f * aw, acy = a1 + 0.5f * ah;
  float pcx = d0 * aw + acx, pcy = d1 * ah + acy;
  float pw = expf(d2) * aw, ph = expf(d3) * ah;
  float mw = (float)(WW * 16 - 1), mh = (float)(HH * 16 - 1);
  boxes[j * 4 + 0] = fminf(fmaxf(pcx - 0.5f * pw, 0.f), mw);
  boxes[j * 4 + 1] = fminf(fmaxf(pcy - 0.5f * ph, 0.f), mh);
  boxes[j * 4 + 2] = fminf(fmaxf(pcx + 0.5f * pw, 0.f), mw);
  boxes[j * 4 + 3] = fminf(fmaxf(pcy + 0.5f * ph, 0.f), mh);
}

// ---------------------------------------------------------------------------
// 4. top-300 selection
// ---------------------------------------------------------------------------
__global__ void __launch_bounds__(1024) topk_kernel(float* __restrict__ scores,
                                                    const float* __restrict__ boxes,
                                                    float* __restrict__ rois)
{
  __shared__ float sv[1024];
  __shared__ int   si[1024];
  const int tid = threadIdx.x;
  for (int t = 0; t < TOPN; ++t) {
    float best = NEGINF; int bi = 0;
    for (int i = tid; i < NANCH; i += 1024) {
      float v = scores[i];
      if (v > best) { best = v; bi = i; }
    }
    sv[tid] = best; si[tid] = bi;
    __syncthreads();
    for (int s = 512; s > 0; s >>= 1) {
      if (tid < s && sv[tid + s] > sv[tid]) { sv[tid] = sv[tid + s]; si[tid] = si[tid + s]; }
      __syncthreads();
    }
    if (tid == 0) {
      int b = si[0];
      rois[t * 4 + 0] = boxes[b * 4 + 0];
      rois[t * 4 + 1] = boxes[b * 4 + 1];
      rois[t * 4 + 2] = boxes[b * 4 + 2];
      rois[t * 4 + 3] = boxes[b * 4 + 3];
      scores[b] = NEGINF;
    }
    __syncthreads();
  }
}

// ---------------------------------------------------------------------------
// 5. ROI align -> pooled bf16 [300][50176]
// ---------------------------------------------------------------------------
__global__ void __launch_bounds__(256) roi_align_kernel(const float* __restrict__ feat,
                                                        const float* __restrict__ rois,
                                                        __bf16* __restrict__ pooled)
{
  __shared__ int xlo[7], xhi[7], ylo[7], yhi[7];
  __shared__ float wx[7], wy[7];
  const int b = blockIdx.x;
  if (threadIdx.x < 7) {
    int p = threadIdx.x;
    float g = (p + 0.5f) / 7.f;
    float x1 = rois[b * 4 + 0] / 16.f, y1 = rois[b * 4 + 1] / 16.f;
    float x2 = rois[b * 4 + 2] / 16.f, y2 = rois[b * 4 + 3] / 16.f;
    float xs = fminf(fmaxf(x1 + (x2 - x1) * g, 0.f), (float)(WW - 1));
    float ys = fminf(fmaxf(y1 + (y2 - y1) * g, 0.f), (float)(HH - 1));
    float x0 = floorf(xs), y0 = floorf(ys);
    xlo[p] = (int)x0; xhi[p] = min((int)x0 + 1, WW - 1); wx[p] = xs - x0;
    ylo[p] = (int)y0; yhi[p] = min((int)y0 + 1, HH - 1); wy[p] = ys - y0;
  }
  __syncthreads();
  for (int t = threadIdx.x; t < KFC; t += 256) {
    int c = t / 49, pp = t % 49, py = pp / 7, px = pp % 7;
    const float* f = feat + c * HWSZ;
    float v00 = f[ylo[py] * WW + xlo[px]];
    float v01 = f[ylo[py] * WW + xhi[px]];
    float v10 = f[yhi[py] * WW + xlo[px]];
    float v11 = f[yhi[py] * WW + xhi[px]];
    float ox = wx[px], oy = wy[py];
    float v = v00 * (1 - oy) * (1 - ox) + v01 * (1 - oy) * ox
            + v10 * oy * (1 - ox) + v11 * oy * ox;
    pooled[(size_t)b * KFC + t] = f2bf(v);
  }
}

// ---------------------------------------------------------------------------
// 6. FC GEMM 300x2048 (K=50176). fc_w (411 MB fp32) is the HBM roofline term:
//    read exactly once, streamed through LDS with double-buffered CDNA5 async
//    loads (global_load_async_to_lds_b128 / s_wait_asynccnt). 8 waves split K,
//    each carries all 19 M-tile accumulators; LDS cross-wave reduce at the end.
// ---------------------------------------------------------------------------
__global__ void __launch_bounds__(256) fc_wmma(const __bf16* __restrict__ Abf, // [300][50176]
                                               const float* __restrict__ Wf,   // [50176][2048]
                                               const float* __restrict__ bias, // [2048]
                                               __bf16* __restrict__ outbf)     // [300][2048]
{
  const int nt = blockIdx.x;                 // 0..127 (16-col tile)
  const int tid = threadIdx.x;
  const int wave = tid >> 5, lane = tid & 31;
  const int half = lane >> 4, l15 = lane & 15;
  const int kw = wave * (KFC / 8);           // 6272 per wave, 196 k-steps

  __shared__ float stage[8][2][32][16];      // per-wave double-buffered B tile
  __shared__ float red[8 * 256];

  const int srow = lane >> 2, scg = lane & 3;   // 8 rows x 4 col-groups per issue
  const uint32_t lds0 =
      (uint32_t)(uintptr_t)&stage[wave][0][0][0] + (uint32_t)(srow * 64 + scg * 16);

  auto issue = [&](int s, int buf) {
    int k0 = kw + s * 32;
    uint32_t dbase = lds0 + (uint32_t)buf * (32 * 16 * 4);
#pragma unroll
    for (int i = 0; i < 4; ++i) {
      int kr = srow + i * 8;
      const float* gp = Wf + (size_t)(k0 + kr) * FCD + nt * 16 + scg * 4;
      uint32_t dst = dbase + (uint32_t)(i * 8 * 64);
      asm volatile("global_load_async_to_lds_b128 %0, %1, off"
                   :: "v"(dst), "v"(gp) : "memory");
    }
  };

  v8f zero = {};
  v8f acc[19];
#pragma unroll
  for (int i = 0; i < 19; ++i) acc[i] = zero;
  const v8bf z8 = {};

  issue(0, 0);
  for (int s = 0; s < 196; ++s) {
    const int buf = s & 1;
    const int k0 = kw + s * 32;
    if (s < 195) {
      issue(s + 1, buf ^ 1);
      asm volatile("s_wait_asynccnt 0x4" ::: "memory");  // previous group landed
    } else {
      asm volatile("s_wait_asynccnt 0x0" ::: "memory");
    }
    v16bf b;
#pragma unroll
    for (int e = 0; e < 16; ++e) b[e] = f2bf(stage[wave][buf][e + 16 * half][l15]);
#pragma unroll
    for (int mt = 0; mt < 19; ++mt) {
      int m = mt * 16 + l15;
      v8bf a0 = z8, a1 = z8;
      if (m < TOPN) {
        const __bf16* ap = Abf + (size_t)m * KFC + k0 + 16 * half;
        a0 = *(const v8bf*)(ap);
        a1 = *(const v8bf*)(ap + 8);
      }
      v16bf a = pack16(a0, a1);
      acc[mt] = __builtin_amdgcn_wmma_f32_16x16x32_bf16(false, a, false, b, (short)0, acc[mt], false, false);
    }
  }

  for (int mt = 0; mt < 19; ++mt) {
#pragma unroll
    for (int r = 0; r < 8; ++r) red[wave * 256 + r * 32 + lane] = acc[mt][r];
    __syncthreads();
    float sum = 0.f;
#pragma unroll
    for (int w = 0; w < 8; ++w) sum += red[w * 256 + tid];
    int rr = tid >> 5, ll = tid & 31;
    int Mg = mt * 16 + rr + 8 * (ll >> 4);
    int Ng = nt * 16 + (ll & 15);
    if (Mg < TOPN) {
      float v = sum + bias[Ng];
      outbf[Mg * FCD + Ng] = f2bf(v > 0.f ? v : 0.f);
    }
    __syncthreads();
  }
}

// ---------------------------------------------------------------------------
// 7. losses
// ---------------------------------------------------------------------------
__global__ void init_acc_kernel(float* acc) {
  if (threadIdx.x < 8) acc[threadIdx.x] = 0.f;
}

__global__ void loss_rpn_ce(const float* __restrict__ cls_s, const int* __restrict__ labels,
                            float* __restrict__ acc)
{
  int j = blockIdx.x * 256 + threadIdx.x;
  float ce = 0.f, mk = 0.f;
  if (j < NANCH) {
    int lab = labels[j];
    if (lab != -1) {
      int a = j / HWSZ, pos = j % HWSZ;
      float s0 = cls_s[a * HWSZ + pos], s1 = cls_s[(AA + a) * HWSZ + pos];
      float mx = fmaxf(s0, s1);
      float lse = mx + logf(expf(s0 - mx) + expf(s1 - mx));
      ce = lse - ((lab == 1) ? s1 : s0);
      mk = 1.f;
    }
  }
  __shared__ float sh1[256], sh2[256];
  sh1[threadIdx.x] = ce; sh2[threadIdx.x] = mk;
  __syncthreads();
  for (int s = 128; s > 0; s >>= 1) {
    if (threadIdx.x < s) { sh1[threadIdx.x] += sh1[threadIdx.x + s]; sh2[threadIdx.x] += sh2[threadIdx.x + s]; }
    __syncthreads();
  }
  if (threadIdx.x == 0) { atomicAdd(&acc[0], sh1[0]); atomicAdd(&acc[1], sh2[0]); }
}

__global__ void loss_rpn_l1(const float* __restrict__ bbox_p,  // [36][9120]
                            const float* __restrict__ tgt,     // [pos][36]
                            const float* __restrict__ biw, const float* __restrict__ bow,
                            float* __restrict__ acc)
{
  int idx = blockIdx.x * 256 + threadIdx.x;
  float L = 0.f;
  if (idx < HWSZ * 36) {
    int pos = idx / 36, c = idx % 36;
    float d = biw[idx] * (bbox_p[c * HWSZ + pos] - tgt[idx]);
    float ad = fabsf(d);
    const float s2 = 9.f;
    float v = (ad < 1.f / s2) ? d * d * (s2 * 0.5f) : (ad - 0.5f / s2);
    L = bow[idx] * v;
  }
  __shared__ float sh[256];
  sh[threadIdx.x] = L;
  __syncthreads();
  for (int s = 128; s > 0; s >>= 1) {
    if (threadIdx.x < s) sh[threadIdx.x] += sh[threadIdx.x + s];
    __syncthreads();
  }
  if (threadIdx.x == 0) atomicAdd(&acc[2], sh[0]);
}

__global__ void loss_roi_ce(const float* __restrict__ cls_score, const int* __restrict__ labels,
                            float* __restrict__ acc)
{
  int i = threadIdx.x;
  float ce = 0.f;
  if (i < TOPN) {
    const float* row = cls_score + i * NCLS;
    float mx = row[0];
    for (int c = 1; c < NCLS; ++c) mx = fmaxf(mx, row[c]);
    float se = 0.f;
    for (int c = 0; c < NCLS; ++c) se += expf(row[c] - mx);
    ce = (mx + logf(se)) - row[labels[i]];
  }
  __shared__ float sh[512];
  sh[i] = ce;
  __syncthreads();
  for (int s = 256; s > 0; s >>= 1) {
    if (i < s) sh[i] += sh[i + s];
    __syncthreads();
  }
  if (i == 0) atomicAdd(&acc[3], sh[0]);
}

__global__ void loss_roi_l1(const float* __restrict__ pred,   // [300][84]
                            const float* __restrict__ tgt,
                            const float* __restrict__ biw, const float* __restrict__ bow,
                            float* __restrict__ acc)
{
  int idx = blockIdx.x * 256 + threadIdx.x;
  float L = 0.f;
  if (idx < TOPN * 84) {
    float d = biw[idx] * (pred[idx] - tgt[idx]);
    float ad = fabsf(d);
    float v = (ad < 1.f) ? 0.5f * d * d : (ad - 0.5f);
    L = bow[idx] * v;
  }
  __shared__ float sh[256];
  sh[threadIdx.x] = L;
  __syncthreads();
  for (int s = 128; s > 0; s >>= 1) {
    if (threadIdx.x < s) sh[threadIdx.x] += sh[threadIdx.x + s];
    __syncthreads();
  }
  if (threadIdx.x == 0) atomicAdd(&acc[4], sh[0]);
}

__global__ void finalize_kernel(const float* __restrict__ acc, float* __restrict__ out) {
  out[0] = acc[0] / fmaxf(acc[1], 1.f) + acc[2] + acc[3] / (float)TOPN + acc[4] / (float)TOPN;
}

// ---------------------------------------------------------------------------
// host launcher
// ---------------------------------------------------------------------------
extern "C" void kernel_launch(void* const* d_in, const int* in_sizes, int n_in,
                              void* d_out, int out_size, void* d_ws, size_t ws_size,
                              hipStream_t stream) {
  const float* net_conv   = (const float*)d_in[0];
  const float* rpn_w      = (const float*)d_in[1];
  const float* rpn_b      = (const float*)d_in[2];
  const float* rpn_cls_w  = (const float*)d_in[3];
  const float* rpn_cls_b  = (const float*)d_in[4];
  const float* rpn_bbox_w = (const float*)d_in[5];
  const float* rpn_bbox_b = (const float*)d_in[6];
  const float* fc_w       = (const float*)d_in[7];
  const float* fc_b       = (const float*)d_in[8];
  const float* cls_w      = (const float*)d_in[9];
  const float* cls_b      = (const float*)d_in[10];
  const float* bbox_w     = (const float*)d_in[11];
  const float* bbox_b     = (const float*)d_in[12];
  const float* anchors    = (const float*)d_in[13];
  const float* rpn_tgt    = (const float*)d_in[14];
  const float* rpn_biw    = (const float*)d_in[15];
  const float* rpn_bow    = (const float*)d_in[16];
  const float* roi_tgt    = (const float*)d_in[17];
  const float* roi_biw    = (const float*)d_in[18];
  const float* roi_bow    = (const float*)d_in[19];
  const int*   rpn_labels = (const int*)d_in[20];
  const int*   roi_labels = (const int*)d_in[21];
  float* out = (float*)d_out;

  char* ws = (char*)d_ws;
  size_t off = 0;
  auto alloc = [&](size_t bytes) -> char* {
    char* p = ws + off;
    off = (off + bytes + 255) & ~(size_t)255;
    return p;
  };
  __bf16* wArs     = (__bf16*)alloc((size_t)9 * CMID * CIN * 2);
  __bf16* featT    = (__bf16*)alloc((size_t)HWSZ * CIN * 2);
  __bf16* rpn_bf   = (__bf16*)alloc((size_t)CMID * HWSZ * 2);
  float*  cls_s    = (float*) alloc((size_t)18 * HWSZ * 4);
  float*  bbox_p   = (float*) alloc((size_t)36 * HWSZ * 4);
  float*  scores   = (float*) alloc((size_t)NANCH * 4);
  float*  boxes    = (float*) alloc((size_t)NANCH * 4 * 4);
  float*  rois     = (float*) alloc((size_t)TOPN * 4 * 4);
  __bf16* pooled   = (__bf16*)alloc((size_t)TOPN * KFC * 2);
  __bf16* fc7      = (__bf16*)alloc((size_t)TOPN * FCD * 2);
  float*  cls_sc   = (float*) alloc((size_t)TOPN * NCLS * 4);
  float*  bbox_pr  = (float*) alloc((size_t)TOPN * 84 * 4);
  float*  acc      = (float*) alloc(8 * 4);

  init_acc_kernel<<<1, 32, 0, stream>>>(acc);

  // RPN conv (bf16 WMMA, 9 shifted GEMMs)
  prep_conv_weights<<<(9 * CMID * CIN + 255) / 256, 256, 0, stream>>>(rpn_w, wArs);
  prep_featT<<<(CIN * HWSZ + 255) / 256, 256, 0, stream>>>(net_conv, featT);
  conv3x3_wmma<<<dim3(HWSZ / 16, CMID / 16), 32, 0, stream>>>(featT, wArs, rpn_b, rpn_bf);

  // 1x1 convs as WMMA GEMMs (A fp32 weights, B bf16 activations, row bias)
  gemm16_wmma<1, 0, 1><<<dim3(HWSZ / 16, 2), 32, 0, stream>>>(
      rpn_cls_w, CMID, rpn_bf, HWSZ, rpn_cls_b, cls_s, HWSZ, 18, HWSZ, CMID);
  gemm16_wmma<1, 0, 1><<<dim3(HWSZ / 16, 3), 32, 0, stream>>>(
      rpn_bbox_w, CMID, rpn_bf, HWSZ, rpn_bbox_b, bbox_p, HWSZ, 36, HWSZ, CMID);

  // proposals + top-300 + ROI align
  proposal_kernel<<<(NANCH + 255) / 256, 256, 0, stream>>>(cls_s, bbox_p, anchors, scores, boxes);
  topk_kernel<<<1, 1024, 0, stream>>>(scores, boxes, rois);
  roi_align_kernel<<<TOPN, 256, 0, stream>>>(net_conv, rois, pooled);

  // FC (fc_w streamed once via async-to-LDS) + heads
  fc_wmma<<<FCD / 16, 256, 0, stream>>>(pooled, fc_w, fc_b, fc7);
  gemm16_wmma<0, 1, 0><<<dim3(2, (TOPN + 15) / 16), 32, 0, stream>>>(
      fc7, FCD, cls_w, NCLS, cls_b, cls_sc, NCLS, TOPN, NCLS, FCD);
  gemm16_wmma<0, 1, 0><<<dim3(6, (TOPN + 15) / 16), 32, 0, stream>>>(
      fc7, FCD, bbox_w, 84, bbox_b, bbox_pr, 84, TOPN, 84, FCD);

  // losses
  loss_rpn_ce<<<(NANCH + 255) / 256, 256, 0, stream>>>(cls_s, rpn_labels, acc);
  loss_rpn_l1<<<(HWSZ * 36 + 255) / 256, 256, 0, stream>>>(bbox_p, rpn_tgt, rpn_biw, rpn_bow, acc);
  loss_roi_ce<<<1, 512, 0, stream>>>(cls_sc, roi_labels, acc);
  loss_roi_l1<<<(TOPN * 84 + 255) / 256, 256, 0, stream>>>(bbox_pr, roi_tgt, roi_biw, roi_bow, acc);
  finalize_kernel<<<1, 1, 0, stream>>>(acc, out);

  (void)in_sizes; (void)n_in; (void)out_size; (void)ws_size;
}
